// InferAMFormer_76433237999713
// MI455X (gfx1250) — matrix operand order repaired and checked by reading, not compiled, not run on hardware
//
#include <hip/hip_runtime.h>
#include <hip/hip_bf16.h>
#include <math.h>

// ---------------------------------------------------------------------------
// Problem constants (from reference): B=4096, N=256, D=64, P=16, NZ=128, K=8
// ---------------------------------------------------------------------------
#define BB 4096
#define NN 256
#define DD 64
#define PP 16
#define NZ 128
#define TOPK 8

typedef __attribute__((ext_vector_type(16))) _Float16 v16h;
typedef __attribute__((ext_vector_type(8)))  _Float16 v8h;
typedef __attribute__((ext_vector_type(8)))  float    v8f;

// ---------------------------------------------------------------------------
// WMMA fragment loaders per CDNA5 ISA 7.12.2 (wave32), vectorized:
// A (16x32 f16, MxK) row-major src: lane holds row M=lane%16; halves h:
//   K = (h&7) + 16*(h>>3) + 8*(lane>=16)  ->  two contiguous 8-half runs.
// B (32x16 f16, KxN) from TRANSPOSED src (N-major): lane holds col N=lane%16;
//   halves h: K = (h&7) + 16*(h>>3) + 8*(lane>=16)  ->  two contiguous 8-half runs.
// C/D (16x16 f32): lane holds col N=lane%16; vgpr r: M = r + 8*(lane>=16).
// ---------------------------------------------------------------------------
__device__ __forceinline__ v16h frag_A(const _Float16* __restrict__ s, int lane, int ld) {
  int row = lane & 15;
  int kb  = (lane >> 4) << 3;
  const _Float16* p = s + row * ld + kb;
  v8h lo = *(const v8h*)(p);
  v8h hi = *(const v8h*)(p + 16);
  return __builtin_shufflevector(lo, hi, 0, 1, 2, 3, 4, 5, 6, 7,
                                         8, 9, 10, 11, 12, 13, 14, 15);
}

__device__ __forceinline__ v16h frag_BT(const _Float16* __restrict__ sT, int lane, int ld) {
  int col = lane & 15;
  int kb  = (lane >> 4) << 3;
  const _Float16* p = sT + col * ld + kb;
  v8h lo = *(const v8h*)(p);
  v8h hi = *(const v8h*)(p + 16);
  return __builtin_shufflevector(lo, hi, 0, 1, 2, 3, 4, 5, 6, 7,
                                         8, 9, 10, 11, 12, 13, 14, 15);
}

__device__ __forceinline__ v8f wmma_f16(v16h a, v16h b, v8f c) {
  return __builtin_amdgcn_wmma_f32_16x16x32_f16(
      /*neg_a=*/false, a, /*neg_b=*/false, b,
      /*c_mod=*/(short)0, c, /*reuse_a=*/false, /*reuse_b=*/false);
}

// ---------------------------------------------------------------------------
// K0a: fp32 -> f16 (straight copy; prompts)
// ---------------------------------------------------------------------------
__global__ void cvt_f16_kernel(const float* __restrict__ src,
                               _Float16* __restrict__ dst, int n) {
  int i = blockIdx.x * blockDim.x + threadIdx.x;
  if (i < n) dst[i] = (_Float16)src[i];
}

// ---------------------------------------------------------------------------
// K0b: fp32 [R,C] -> f16 transposed [C,R] (weights; one-time, L2-resident after)
// ---------------------------------------------------------------------------
__global__ void cvt_f16_t_kernel(const float* __restrict__ src,
                                 _Float16* __restrict__ dst, int R, int C) {
  int i = blockIdx.x * blockDim.x + threadIdx.x;
  if (i >= R * C) return;
  int c = i / R, r = i - c * R;  // output-coalesced
  dst[(size_t)c * R + r] = (_Float16)src[(size_t)r * C + c];
}

// ---------------------------------------------------------------------------
// K1: z-score the first NZ columns; emit fp32 (residual) and f16 (GEMM A).
// ---------------------------------------------------------------------------
__global__ void zscore_kernel(const float* __restrict__ x,
                              const float* __restrict__ mean,
                              const float* __restrict__ stdv,
                              float* __restrict__ xz,
                              _Float16* __restrict__ xz_h) {
  int i = blockIdx.x * blockDim.x + threadIdx.x;
  if (i >= BB * NN) return;
  int j = i & (NN - 1);
  float v = x[i];
  if (j < NZ) v = (v - mean[j]) / stdv[j];
  xz[i]   = v;
  xz_h[i] = (_Float16)v;
}

// ---------------------------------------------------------------------------
// K2: fused tok-embedding GEMM + 4 projections + prompt score GEMMs.
// One workgroup = (16 batch rows) x (1 token). 128 threads = 4 waves.
//   emb   = xz[16,256] @ tokW[:, n*64:(n+1)*64] + tok_b        (WMMA, K=256)
//   k/v_a = emb @ {kA,vA}+b ; k/v_m = log(relu(emb)+1e-5) @ {kM,vM}+b
//   score = prompt[16,64] @ k^T / 8 -> scores[B,P,N]
// emb never touches HBM; all B operands pre-transposed -> contiguous frags.
// ---------------------------------------------------------------------------
__global__ __launch_bounds__(128)
void emb_proj_kernel(const _Float16* __restrict__ xz_h,     // [B,256]
                     const _Float16* __restrict__ tokW_T,   // [16384,256] (n-major)
                     const float*    __restrict__ tok_b,    // [16384]
                     const _Float16* __restrict__ projW_T,  // [4][64n][64k] kA,vA,kM,vM
                     const float* __restrict__ kAb, const float* __restrict__ vAb,
                     const float* __restrict__ kMb, const float* __restrict__ vMb,
                     const _Float16* __restrict__ prompts_h,// [2][16][64]
                     float* __restrict__ scores_add,        // [B,16,256]
                     float* __restrict__ scores_mul,
                     _Float16* __restrict__ v_add,          // [B,256,64]
                     _Float16* __restrict__ v_mul) {
  __shared__ __align__(16) _Float16 sEmb[16 * 64];
  __shared__ __align__(16) _Float16 sLog[16 * 64];
  __shared__ __align__(16) _Float16 sK[16 * 64];

  const int n    = blockIdx.x;       // token
  const int b0   = blockIdx.y * 16;  // batch tile
  const int lane = threadIdx.x & 31;
  const int w    = threadIdx.x >> 5;

  // ---- Phase 1: emb tile [16 x 64]; wave w owns cols 16w..16w+15 ----
  const _Float16* Ar = xz_h + (size_t)b0 * NN;
  const _Float16* Bt = tokW_T + ((size_t)n * DD + 16 * w) * NN;
  v8f acc = {};
#pragma unroll
  for (int k0 = 0; k0 < NN; k0 += 32)
    acc = wmma_f16(frag_A(Ar + k0, lane, NN), frag_BT(Bt + k0, lane, NN), acc);
  {
    int c  = 16 * w + (lane & 15);
    int rb = (lane >> 4) << 3;
    float bias = tok_b[n * DD + c];
#pragma unroll
    for (int r = 0; r < 8; ++r) {
      float e = acc[r] + bias;
      sEmb[(rb + r) * 64 + c] = (_Float16)e;
      sLog[(rb + r) * 64 + c] = (_Float16)__logf(fmaxf(e, 0.f) + 1e-5f);
    }
  }
  __syncthreads();

  // ---- Phase 2: branches (0 = additive, 1 = multiplicative/log) ----
  for (int br = 0; br < 2; ++br) {
    const _Float16* sX   = br ? sLog : sEmb;
    const _Float16* kW_T = projW_T + (size_t)(2 * br + 0) * 4096 + (16 * w) * 64;
    const _Float16* vW_T = projW_T + (size_t)(2 * br + 1) * 4096 + (16 * w) * 64;
    const float* kb = br ? kMb : kAb;
    const float* vb = br ? vMb : vAb;

    // k tile -> LDS (f16, row-major [16 rows][64 d])
    v8f ak = {};
#pragma unroll
    for (int kc = 0; kc < 64; kc += 32)
      ak = wmma_f16(frag_A(sX + kc, lane, 64), frag_BT(kW_T + kc, lane, 64), ak);
    {
      int c = 16 * w + (lane & 15), rb = (lane >> 4) << 3;
      float bias = kb[c];
#pragma unroll
      for (int r = 0; r < 8; ++r) sK[(rb + r) * 64 + c] = (_Float16)(ak[r] + bias);
    }

    // v tile -> global (f16)
    v8f av = {};
#pragma unroll
    for (int kc = 0; kc < 64; kc += 32)
      av = wmma_f16(frag_A(sX + kc, lane, 64), frag_BT(vW_T + kc, lane, 64), av);
    {
      int c = 16 * w + (lane & 15), rb = (lane >> 4) << 3;
      float bias = vb[c];
      _Float16* vd = br ? v_mul : v_add;
#pragma unroll
      for (int r = 0; r < 8; ++r)
        vd[((size_t)(b0 + rb + r) * NN + n) * DD + c] = (_Float16)(av[r] + bias);
    }
    __syncthreads();  // sK fully written before score read

    // scores = prompt[16,64] @ sK^T. sK is [batch j][d] == B^T layout already.
    // Wave 0 only (wave-uniform branch; EXEC all-1 inside).
    if (w == 0) {
      v8f as = {};
#pragma unroll
      for (int kc = 0; kc < 64; kc += 32)
        as = wmma_f16(frag_A(prompts_h + br * (PP * DD) + kc, lane, 64),
                      frag_BT(sK + kc, lane, 64), as);
      float* sd = br ? scores_mul : scores_add;
      int j = lane & 15, rb = (lane >> 4) << 3;
#pragma unroll
      for (int r = 0; r < 8; ++r)
        sd[((size_t)(b0 + j) * PP + (rb + r)) * NN + n] = as[r] * 0.125f;  // /sqrt(64)
    }
    __syncthreads();  // before sK reuse
  }
}

// ---------------------------------------------------------------------------
// K3: per (b,p) top-8 over 256 scores, softmax, weighted V gather -> cat (f16)
// ---------------------------------------------------------------------------
__global__ void topk_attn_kernel(const float* __restrict__ scores,  // [B,P,N]
                                 const _Float16* __restrict__ v,    // [B,N,D]
                                 _Float16* __restrict__ cat,        // [B,2048]
                                 int catOffset, int isMul) {
  int t = blockIdx.x * blockDim.x + threadIdx.x;
  if (t >= BB * PP) return;
  int b = t >> 4, p = t & 15;
  const float* sr = scores + (size_t)t * NN;

  unsigned chosen[NN / 32];
#pragma unroll
  for (int i = 0; i < NN / 32; ++i) chosen[i] = 0u;

  float val[TOPK];
  int   idx[TOPK];
  for (int s = 0; s < TOPK; ++s) {
    float best = -INFINITY;
    int bj = 0;
    for (int j = 0; j < NN; ++j) {
      if ((chosen[j >> 5] >> (j & 31)) & 1u) continue;
      float sv = sr[j];
      if (sv > best) { best = sv; bj = j; }
    }
    chosen[bj >> 5] |= 1u << (bj & 31);
    val[s] = best;
    idx[s] = bj;
  }
  float m = val[0];  // first pick is the max
  float den = 0.f;
#pragma unroll
  for (int s = 0; s < TOPK; ++s) { val[s] = __expf(val[s] - m); den += val[s]; }
  float rden = 1.f / den;
#pragma unroll
  for (int s = 0; s < TOPK; ++s) val[s] *= rden;

  _Float16* dst = cat + (size_t)b * (2 * PP * DD) + catOffset + p * DD;
  // accumulate in 16-wide chunks to keep register pressure low
  for (int d0 = 0; d0 < DD; d0 += 16) {
    float o[16];
#pragma unroll
    for (int i = 0; i < 16; ++i) o[i] = 0.f;
    for (int s = 0; s < TOPK; ++s) {
      const _Float16* vr = v + ((size_t)b * NN + idx[s]) * DD + d0;
      float wg = val[s];
#pragma unroll
      for (int i = 0; i < 16; ++i) o[i] += wg * (float)vr[i];
    }
#pragma unroll
    for (int i = 0; i < 16; ++i) {
      float ov = o[i];
      if (isMul) ov = __expf(fminf(ov, 10.f));  // clamp(max=10) then exp
      dst[d0 + i] = (_Float16)ov;
    }
  }
}

// ---------------------------------------------------------------------------
// K4: generic f16 GEMM  C[M,Nc] = A[M,K] @ W[K,Nc] + bias  (fp32 out, WMMA)
// LDS-free. Each wave owns a 32x32 output tile: 2 A-frags x 2 B-frags -> 4
// WMMAs per K-chunk from 4 fragment loads (2x register reuse vs 1 tile/wave).
// Workgroup = 4 waves side-by-side = 32x128 tile. Grid: (M/32, Nc/128).
// ---------------------------------------------------------------------------
__global__ __launch_bounds__(128)
void gemm_f16_kernel(const _Float16* __restrict__ A,
                     const _Float16* __restrict__ WT,  // [Nc,K]
                     const float* __restrict__ bias, float* __restrict__ C,
                     int M, int K, int Nc) {
  const int lane = threadIdx.x & 31, w = threadIdx.x >> 5;
  const int m0 = blockIdx.x * 32;
  const int n0 = blockIdx.y * 128 + w * 32;
  const _Float16* Ar0 = A + (size_t)m0 * K;
  const _Float16* Ar1 = A + (size_t)(m0 + 16) * K;
  const _Float16* Bt0 = WT + (size_t)n0 * K;
  const _Float16* Bt1 = WT + (size_t)(n0 + 16) * K;

  v8f acc00 = {}, acc01 = {}, acc10 = {}, acc11 = {};
  for (int k0 = 0; k0 < K; k0 += 32) {
    v16h a0 = frag_A(Ar0 + k0, lane, K);
    v16h a1 = frag_A(Ar1 + k0, lane, K);
    v16h b0 = frag_BT(Bt0 + k0, lane, K);
    v16h b1 = frag_BT(Bt1 + k0, lane, K);
    acc00 = wmma_f16(a0, b0, acc00);
    acc01 = wmma_f16(a0, b1, acc01);
    acc10 = wmma_f16(a1, b0, acc10);
    acc11 = wmma_f16(a1, b1, acc11);
  }

  int cl = lane & 15;
  int rb = (lane >> 4) << 3;
  float bv0 = bias[n0 + cl];
  float bv1 = bias[n0 + 16 + cl];
#pragma unroll
  for (int r = 0; r < 8; ++r) {
    size_t row0 = (size_t)(m0 + rb + r) * Nc;
    size_t row1 = (size_t)(m0 + 16 + rb + r) * Nc;
    C[row0 + n0 + cl]      = acc00[r] + bv0;
    C[row0 + n0 + 16 + cl] = acc01[r] + bv1;
    C[row1 + n0 + cl]      = acc10[r] + bv0;
    C[row1 + n0 + 16 + cl] = acc11[r] + bv1;
  }
}

// ---------------------------------------------------------------------------
// K5: y = tanhshrink(g); LN(y)*gamma+beta; optional residual: out = res + ln*s
// One block (256 threads) per row. R in {256, 1024}.
// ---------------------------------------------------------------------------
__global__ void tanhshrink_ln_kernel(const float* __restrict__ g,
                                     const float* __restrict__ gamma,
                                     const float* __restrict__ beta,
                                     const float* __restrict__ resid, float resScale,
                                     _Float16* __restrict__ out_h,
                                     float* __restrict__ out_f, int R) {
  __shared__ float sy[1024];
  __shared__ float rs[256], rss[256];
  const int row = blockIdx.x, tid = threadIdx.x;
  const float* gr = g + (size_t)row * R;
  float s = 0.f, ss = 0.f;
  for (int i = tid; i < R; i += 256) {
    float x = gr[i];
    float y = x - tanhf(x);
    sy[i] = y;
    s += y;
    ss += y * y;
  }
  rs[tid] = s;
  rss[tid] = ss;
  __syncthreads();
  for (int off = 128; off > 0; off >>= 1) {
    if (tid < off) { rs[tid] += rs[tid + off]; rss[tid] += rss[tid + off]; }
    __syncthreads();
  }
  float mean = rs[0] / (float)R;
  float var  = rss[0] / (float)R - mean * mean;  // biased, matches torch LN
  float inv  = rsqrtf(var + 1e-5f);
  for (int i = tid; i < R; i += 256) {
    float yn = (sy[i] - mean) * inv * gamma[i] + beta[i];
    if (resid) yn = resid[(size_t)row * R + i] + yn * resScale;
    if (out_h) out_h[(size_t)row * R + i] = (_Float16)yn;
    if (out_f) out_f[(size_t)row * R + i] = yn;
  }
}

// ---------------------------------------------------------------------------
// K6: head: log_hz = h2 @ W3 + b3; searchsorted(base_event_times, 10, right);
//     out[b] = 1 - exp(-base_hazards[idx] * exp(log_hz)). One wave per row.
// ---------------------------------------------------------------------------
__global__ void head_kernel(const float* __restrict__ h2, const float* __restrict__ W3,
                            const float* __restrict__ b3,
                            const float* __restrict__ bet,   // [101]
                            const float* __restrict__ bh,    // [101]
                            float* __restrict__ out) {
  int wid  = (blockIdx.x * blockDim.x + threadIdx.x) >> 5;
  int lane = threadIdx.x & 31;
  if (wid >= BB) return;
  const float* hr = h2 + (size_t)wid * 1024;
  float s = 0.f;
  for (int i = lane; i < 1024; i += 32) s += hr[i] * W3[i];
#pragma unroll
  for (int off = 16; off > 0; off >>= 1) s += __shfl_down(s, off, 32);
  if (lane == 0) {
    float log_hz = s + b3[0];
    int cnt = 0;
    for (int i = 0; i < 101; ++i) cnt += (bet[i] <= 10.f) ? 1 : 0;  // side='right'
    int idx = cnt - 1;
    idx = idx < 0 ? 0 : (idx > 100 ? 100 : idx);
    out[wid] = 1.f - __expf(-bh[idx] * __expf(log_hz));
  }
}

// ---------------------------------------------------------------------------
// Host-side orchestration
// ---------------------------------------------------------------------------
static inline size_t ws_take(size_t& off, size_t bytes) {
  off = (off + 255) & ~(size_t)255;
  size_t p = off;
  off += bytes;
  return p;
}

extern "C" void kernel_launch(void* const* d_in, const int* in_sizes, int n_in,
                              void* d_out, int out_size, void* d_ws, size_t ws_size,
                              hipStream_t stream) {
  (void)in_sizes; (void)n_in; (void)out_size; (void)ws_size;
  const float* x      = (const float*)d_in[0];
  const float* mean   = (const float*)d_in[1];
  const float* stdv   = (const float*)d_in[2];
  const float* bet    = (const float*)d_in[3];
  const float* bh     = (const float*)d_in[4];
  const float* tok_W  = (const float*)d_in[5];
  const float* tok_b  = (const float*)d_in[6];
  const float* pr_add = (const float*)d_in[7];
  const float* pr_mul = (const float*)d_in[8];
  const float* kA_W   = (const float*)d_in[9];
  const float* kA_b   = (const float*)d_in[10];
  const float* vA_W   = (const float*)d_in[11];
  const float* vA_b   = (const float*)d_in[12];
  const float* kM_W   = (const float*)d_in[13];
  const float* kM_b   = (const float*)d_in[14];
  const float* vM_W   = (const float*)d_in[15];
  const float* vM_b   = (const float*)d_in[16];
  const float* ffn_W  = (const float*)d_in[17];
  const float* ffn_b  = (const float*)d_in[18];
  const float* ln0_g  = (const float*)d_in[19];
  const float* ln0_b  = (const float*)d_in[20];
  const float* W1     = (const float*)d_in[21];
  const float* b1     = (const float*)d_in[22];
  const float* ln1_g  = (const float*)d_in[23];
  const float* ln1_b  = (const float*)d_in[24];
  const float* W2     = (const float*)d_in[25];
  const float* b2     = (const float*)d_in[26];
  const float* ln2_g  = (const float*)d_in[27];
  const float* ln2_b  = (const float*)d_in[28];
  const float* W3     = (const float*)d_in[29];
  const float* b3     = (const float*)d_in[30];
  float* out = (float*)d_out;

  uint8_t* ws = (uint8_t*)d_ws;
  size_t off = 0;
  float*     xz       = (float*)    (ws + ws_take(off, (size_t)BB * NN * 4));
  _Float16*  xz_h     = (_Float16*) (ws + ws_take(off, (size_t)BB * NN * 2));
  _Float16*  tokW_T   = (_Float16*) (ws + ws_take(off, (size_t)NN * NN * DD * 2));
  _Float16*  projW_T  = (_Float16*) (ws + ws_take(off, (size_t)4 * DD * DD * 2));
  _Float16*  prompt_h = (_Float16*) (ws + ws_take(off, (size_t)2 * PP * DD * 2));
  _Float16*  ffnW_T   = (_Float16*) (ws + ws_take(off, (size_t)2 * PP * DD * NN * 2));
  _Float16*  W1_T     = (_Float16*) (ws + ws_take(off, (size_t)NN * 1024 * 2));
  _Float16*  W2_T     = (_Float16*) (ws + ws_take(off, (size_t)1024 * 1024 * 2));
  float*     sc_add   = (float*)    (ws + ws_take(off, (size_t)BB * PP * NN * 4));
  float*     sc_mul   = (float*)    (ws + ws_take(off, (size_t)BB * PP * NN * 4));
  _Float16*  v_add    = (_Float16*) (ws + ws_take(off, (size_t)BB * NN * DD * 2));
  _Float16*  v_mul    = (_Float16*) (ws + ws_take(off, (size_t)BB * NN * DD * 2));
  _Float16*  cat_h    = (_Float16*) (ws + ws_take(off, (size_t)BB * 2 * PP * DD * 2));
  float*     g0       = (float*)    (ws + ws_take(off, (size_t)BB * NN * 4));
  _Float16*  h_h      = (_Float16*) (ws + ws_take(off, (size_t)BB * NN * 2));
  float*     g1       = (float*)    (ws + ws_take(off, (size_t)BB * 1024 * 4));
  _Float16*  h1_h     = (_Float16*) (ws + ws_take(off, (size_t)BB * 1024 * 2));
  float*     g2       = (float*)    (ws + ws_take(off, (size_t)BB * 1024 * 4));
  float*     h2       = g1;  // g1 dead after ln1 -> reuse for h2

  auto cvt = [&](const float* s, _Float16* d, int n) {
    cvt_f16_kernel<<<(n + 255) / 256, 256, 0, stream>>>(s, d, n);
  };
  auto cvtT = [&](const float* s, _Float16* d, int R, int C) {
    cvt_f16_t_kernel<<<(R * C + 255) / 256, 256, 0, stream>>>(s, d, R, C);
  };

  // ---- weight conversions fp32 -> f16 (B operands transposed to [Nc,K]) ----
  cvtT(tok_W, tokW_T, NN, NN * DD);
  cvtT(kA_W, projW_T + 0 * DD * DD, DD, DD);
  cvtT(vA_W, projW_T + 1 * DD * DD, DD, DD);
  cvtT(kM_W, projW_T + 2 * DD * DD, DD, DD);
  cvtT(vM_W, projW_T + 3 * DD * DD, DD, DD);
  cvt(pr_add, prompt_h + 0,       PP * DD);
  cvt(pr_mul, prompt_h + PP * DD, PP * DD);
  cvtT(ffn_W, ffnW_T, 2 * PP * DD, NN);
  cvtT(W1, W1_T, NN, 1024);
  cvtT(W2, W2_T, 1024, 1024);

  // ---- z-score ----
  zscore_kernel<<<(BB * NN + 255) / 256, 256, 0, stream>>>(x, mean, stdv, xz, xz_h);

  // ---- fused tok GEMM + projections + scores (WMMA heavy) ----
  emb_proj_kernel<<<dim3(NN, BB / 16), 128, 0, stream>>>(
      xz_h, tokW_T, tok_b, projW_T, kA_b, vA_b, kM_b, vM_b, prompt_h,
      sc_add, sc_mul, v_add, v_mul);

  // ---- top-k attention, both branches -> cat ----
  topk_attn_kernel<<<(BB * PP + 255) / 256, 256, 0, stream>>>(
      sc_add, v_add, cat_h, /*catOffset=*/0, /*isMul=*/0);
  topk_attn_kernel<<<(BB * PP + 255) / 256, 256, 0, stream>>>(
      sc_mul, v_mul, cat_h, /*catOffset=*/PP * DD, /*isMul=*/1);

  // ---- ffn: cat[4096,2048] @ ffn_W[2048,256] ----
  gemm_f16_kernel<<<dim3(BB / 32, NN / 128), 128, 0, stream>>>(
      cat_h, ffnW_T, ffn_b, g0, BB, 2 * PP * DD, NN);
  // flat = LN(tanhshrink(g0)); h = xz + 0.01*flat
  tanhshrink_ln_kernel<<<BB, 256, 0, stream>>>(g0, ln0_g, ln0_b, xz, 0.01f,
                                               h_h, (float*)nullptr, NN);

  // ---- h @ W1[256,1024]; LN ----
  gemm_f16_kernel<<<dim3(BB / 32, 1024 / 128), 128, 0, stream>>>(
      h_h, W1_T, b1, g1, BB, NN, 1024);
  tanhshrink_ln_kernel<<<BB, 256, 0, stream>>>(g1, ln1_g, ln1_b, (const float*)nullptr,
                                               0.f, h1_h, (float*)nullptr, 1024);

  // ---- h1 @ W2[1024,1024]; LN (fp32 out for head) ----
  gemm_f16_kernel<<<dim3(BB / 32, 1024 / 128), 128, 0, stream>>>(
      h1_h, W2_T, b2, g2, BB, 1024, 1024);
  tanhshrink_ln_kernel<<<BB, 256, 0, stream>>>(g2, ln2_g, ln2_b, (const float*)nullptr,
                                               0.f, (_Float16*)nullptr, h2, 1024);

  // ---- head: dot + hazard transform + searchsorted ----
  head_kernel<<<(BB * 32 + 255) / 256, 256, 0, stream>>>(h2, W3, b3, bet, bh, out);
}
